// FirstStage_33294586479280
// MI455X (gfx1250) — compile-verified, matrix operand order checked by reading, not run on hardware
//
#include <hip/hip_runtime.h>
#include <hip/hip_bf16.h>

// ---------------------------------------------------------------------------
// DA-RNN stage-1 encoder for MI455X (gfx1250, wave32, WMMA bf16).
// B=2048, T=64 (63 steps), N_IN=128, H=128.
// ---------------------------------------------------------------------------

typedef __attribute__((ext_vector_type(16))) __bf16 v16bf;
typedef __attribute__((ext_vector_type(8)))  float  v8f;

#define Bsz   2048
#define TT    64
#define STEPS 63
#define NIN   128
#define HH    128
#define RPAD  136          // padded bf16 row stride: 68 dwords == 4 mod 64 banks

#define LOG2E      1.4426950408889634f
#define TWO_LOG2E  2.8853900817779268f

__device__ __forceinline__ unsigned short f32_to_bf16(float f) {
    unsigned int u = __float_as_uint(f);
    unsigned int r = 0x7FFFu + ((u >> 16) & 1u);   // round-to-nearest-even
    u += r;
    return (unsigned short)(u >> 16);
}

__device__ __forceinline__ float fast_tanh(float x) {
    // tanh(x) = 1 - 2/(exp2(2x*log2e)+1); saturates to +-1 at +-inf
    float e = __builtin_amdgcn_exp2f(x * TWO_LOG2E);
    return 1.0f - 2.0f * __builtin_amdgcn_rcpf(e + 1.0f);
}

__device__ __forceinline__ float fast_sigmoid(float x) {
    float e = __builtin_amdgcn_exp2f(-x * LOG2E);
    return __builtin_amdgcn_rcpf(1.0f + e);
}

// CDNA5 16-bit A/B fragment (lane, element j) -> k index within a 32-wide K tile
__device__ __forceinline__ int frag_k(int hi, int j) {
    int v = j >> 1;
    return ((v & 4) ? 16 : 0) + hi * 8 + (v & 3) * 2 + (j & 1);
}

// Load a 16x32 bf16 A-fragment from a padded row-major bf16 LDS array.
// k-map is table order (frag_k): elems 0..7 -> k0+hi*8.. ; elems 8..15 -> k0+16+hi*8..
__device__ __forceinline__ v16bf load_a_frag(const unsigned short* row, int hi) {
    union { v16bf v; uint4 q[2]; } a;
    a.q[0] = *(const uint4*)(row + hi * 8);
    a.q[1] = *(const uint4*)(row + 16 + hi * 8);
    return a.v;
}

// ---------------------------------------------------------------------------
// Kernel 0: repack weights into bf16 fragment-ordered buffers.
// B[k][n] = W[n][k]; fragment f = nt*NKT + kt, elem idx = ((f*32+L)*16+j).
// Regions: Wh (NT=4, NKT=8, K=256) | W_ih (NT=32, NKT=4, K=128) | W_hh (same).
// ---------------------------------------------------------------------------
__global__ void prep_weights_kernel(const float* __restrict__ Wih,
                                    const float* __restrict__ Whh,
                                    const float* __restrict__ Wh,
                                    unsigned short* __restrict__ bwh,
                                    unsigned short* __restrict__ bih,
                                    unsigned short* __restrict__ bhh) {
    int g = blockIdx.x * blockDim.x + threadIdx.x;   // 147456 total
    const float* src;
    unsigned short* dst;
    int NKT, Kdim, l;
    if (g < 16384)        { src = Wh;  dst = bwh; NKT = 8; Kdim = 256; l = g; }
    else if (g < 81920)   { src = Wih; dst = bih; NKT = 4; Kdim = 128; l = g - 16384; }
    else if (g < 147456)  { src = Whh; dst = bhh; NKT = 4; Kdim = 128; l = g - 81920; }
    else return;

    int j  = l & 15;
    int L  = (l >> 4) & 31;
    int f  = l >> 9;
    int kt = f % NKT;
    int nt = f / NKT;
    int n  = nt * 16 + (L & 15);
    int k  = kt * 32 + frag_k(L >> 4, j);
    dst[l] = f32_to_bf16(src[n * Kdim + k]);
}

// ---------------------------------------------------------------------------
// Kernel 1: x_proj[m=(b*128+n)][T] = sum_t input[b][t][n] * Wi[T][t] + bi[T]
// ---------------------------------------------------------------------------
__global__ void xproj_kernel(const float* __restrict__ input,
                             const float* __restrict__ Wi,
                             const float* __restrict__ bi,
                             float* __restrict__ xproj) {
    __shared__ float wil[64 * 63];
    __shared__ float al[4 * 63];
    __shared__ float bil[64];

    int tid = threadIdx.x;
    int m0  = blockIdx.x * 4;

    for (int i = tid; i < 64 * 63; i += 256) wil[i] = Wi[i];
    if (tid < 64) bil[tid] = bi[tid];
    for (int i = tid; i < 4 * 63; i += 256) {
        int r = i / 63, k = i - r * 63;
        int m = m0 + r;
        int b = m >> 7, n = m & 127;
        al[i] = input[((size_t)b * 63 + k) * 128 + n];
    }
    __syncthreads();

    int r   = tid >> 6;
    int col = tid & 63;
    float acc = bil[col];
#pragma unroll 7
    for (int k = 0; k < 63; ++k)
        acc = fmaf(al[r * 63 + k], wil[col * 63 + k], acc);
    xproj[(size_t)(m0 + r) * 64 + col] = acc;
}

// ---------------------------------------------------------------------------
// Kernel 2: the recurrent scan. One block = 16 batch rows, 256 threads = 8
// waves, 63 timesteps. State lives in LDS (bf16 shadows for WMMA A-operands);
// GEMMs use v_wmma_f32_16x16x32_bf16 with f32 accumulate.
// ---------------------------------------------------------------------------
__global__ __launch_bounds__(256, 1)
void scan_kernel(const float* __restrict__ input,
                 const float* __restrict__ b_ih,
                 const float* __restrict__ b_hh,
                 const float* __restrict__ bh,
                 const float* __restrict__ Wa,
                 const float* __restrict__ xproj,
                 const unsigned short* __restrict__ bwh,
                 const unsigned short* __restrict__ bih,
                 const unsigned short* __restrict__ bhh,
                 float* __restrict__ out_w,
                 float* __restrict__ out_h) {
    __shared__ float          sh_c[16 * 128];       // f32 cell state (recurrence precision)
    __shared__ unsigned short sh_hbf[16 * RPAD];    // bf16 h, padded rows
    __shared__ unsigned short sh_cbf[16 * RPAD];    // bf16 c, padded rows
    __shared__ unsigned short sh_wxbf[16 * RPAD];   // bf16 wx, padded rows
    __shared__ float sh_score[16 * 64];
    __shared__ float sh_attn[16 * 128];
    __shared__ float sh_gates[16 * 512];
    __shared__ float sh_bh[64];
    __shared__ float sh_wa[64];
    __shared__ float sh_bias[512];

    const int tid  = threadIdx.x;
    const int wave = tid >> 5;
    const int lane = tid & 31;
    const int hi   = lane >> 4;
    const int nl   = lane & 15;
    const int brow = blockIdx.x * 16;

    // init state + constants
    for (int i = tid; i < 16 * 128; i += 256) sh_c[i] = 0.f;
    for (int i = tid; i < 16 * RPAD; i += 256) {
        sh_hbf[i] = 0; sh_cbf[i] = 0; sh_wxbf[i] = 0;
    }
    if (tid < 64) { sh_bh[tid] = bh[tid]; sh_wa[tid] = Wa[tid]; }
    for (int i = tid; i < 512; i += 256) sh_bias[i] = b_ih[i] + b_hh[i];
    __syncthreads();

    for (int t = 0; t < STEPS; ++t) {
        // ---- Phase 1: score[16,64] = [h,c] @ Wh^T + bh  (waves 0..3) ------
        if (wave < 4) {
            const int nt = wave;
            v8f acc = {0.f, 0.f, 0.f, 0.f, 0.f, 0.f, 0.f, 0.f};
#pragma unroll
            for (int kt = 0; kt < 8; ++kt) {
                const unsigned short* row = (kt < 4)
                    ? (sh_hbf + nl * RPAD + kt * 32)
                    : (sh_cbf + nl * RPAD + (kt - 4) * 32);
                v16bf av = load_a_frag(row, hi);
                v16bf bf = *(const v16bf*)(bwh + ((size_t)(nt * 8 + kt) * 32 + lane) * 16);
                acc = __builtin_amdgcn_wmma_f32_16x16x32_bf16(
                        false, av, false, bf, (short)0, acc, false, false);
            }
            const int n = nt * 16 + nl;
#pragma unroll
            for (int i = 0; i < 8; ++i)
                sh_score[(hi * 8 + i) * 64 + n] = acc[i] + sh_bh[n];
        }
        __syncthreads();

        // ---- Phase 2: attn[m][n] = sum_T Wa[T]*tanh(score[m][T]+xproj) ----
        //      (ba dropped: softmax is shift-invariant)
#pragma unroll
        for (int it = 0; it < 8; ++it) {
            int p = it * 256 + tid;
            int m = p >> 7, n = p & 127;
            const float* xp = xproj + ((size_t)(brow + m) * 128 + n) * 64;
            float acc = 0.f;
#pragma unroll 4
            for (int T = 0; T < 64; T += 4) {
                float4 x4 = *(const float4*)(xp + T);
                acc = fmaf(sh_wa[T + 0], fast_tanh(sh_score[m * 64 + T + 0] + x4.x), acc);
                acc = fmaf(sh_wa[T + 1], fast_tanh(sh_score[m * 64 + T + 1] + x4.y), acc);
                acc = fmaf(sh_wa[T + 2], fast_tanh(sh_score[m * 64 + T + 2] + x4.z), acc);
                acc = fmaf(sh_wa[T + 3], fast_tanh(sh_score[m * 64 + T + 3] + x4.w), acc);
            }
            sh_attn[m * 128 + n] = acc;
        }
        __syncthreads();

        // ---- Phase 3: softmax over n, wx = alpha * x_t, emit out_w --------
#pragma unroll
        for (int rr = 0; rr < 2; ++rr) {
            int m = wave * 2 + rr;
            float v0 = sh_attn[m * 128 + lane];
            float v1 = sh_attn[m * 128 + lane + 32];
            float v2 = sh_attn[m * 128 + lane + 64];
            float v3 = sh_attn[m * 128 + lane + 96];
            float mx = fmaxf(fmaxf(v0, v1), fmaxf(v2, v3));
#pragma unroll
            for (int off = 16; off > 0; off >>= 1)
                mx = fmaxf(mx, __shfl_xor(mx, off, 32));
            float e0 = __builtin_amdgcn_exp2f((v0 - mx) * LOG2E);
            float e1 = __builtin_amdgcn_exp2f((v1 - mx) * LOG2E);
            float e2 = __builtin_amdgcn_exp2f((v2 - mx) * LOG2E);
            float e3 = __builtin_amdgcn_exp2f((v3 - mx) * LOG2E);
            float s = e0 + e1 + e2 + e3;
#pragma unroll
            for (int off = 16; off > 0; off >>= 1)
                s += __shfl_xor(s, off, 32);
            float inv = 1.0f / s;
            const float* xin = input + ((size_t)(brow + m) * 63 + t) * 128;
            float*       ow  = out_w + ((size_t)(brow + m) * 63 + t) * 128;
            float w0 = e0 * inv * xin[lane];
            float w1 = e1 * inv * xin[lane + 32];
            float w2 = e2 * inv * xin[lane + 64];
            float w3 = e3 * inv * xin[lane + 96];
            ow[lane]      = w0;  sh_wxbf[m * RPAD + lane]      = f32_to_bf16(w0);
            ow[lane + 32] = w1;  sh_wxbf[m * RPAD + lane + 32] = f32_to_bf16(w1);
            ow[lane + 64] = w2;  sh_wxbf[m * RPAD + lane + 64] = f32_to_bf16(w2);
            ow[lane + 96] = w3;  sh_wxbf[m * RPAD + lane + 96] = f32_to_bf16(w3);
        }
        __syncthreads();

        // ---- Phase 4: gates[16,512] = wx@W_ih^T + h@W_hh^T + bias ---------
        {
            v8f acc0 = {0.f,0.f,0.f,0.f,0.f,0.f,0.f,0.f};
            v8f acc1 = acc0, acc2 = acc0, acc3 = acc0;
#pragma unroll
            for (int kt = 0; kt < 4; ++kt) {           // wx contribution
                v16bf av = load_a_frag(sh_wxbf + nl * RPAD + kt * 32, hi);
                v16bf b0 = *(const v16bf*)(bih + ((size_t)((wave     ) * 4 + kt) * 32 + lane) * 16);
                v16bf b1 = *(const v16bf*)(bih + ((size_t)((wave +  8) * 4 + kt) * 32 + lane) * 16);
                v16bf b2 = *(const v16bf*)(bih + ((size_t)((wave + 16) * 4 + kt) * 32 + lane) * 16);
                v16bf b3 = *(const v16bf*)(bih + ((size_t)((wave + 24) * 4 + kt) * 32 + lane) * 16);
                acc0 = __builtin_amdgcn_wmma_f32_16x16x32_bf16(false, av, false, b0, (short)0, acc0, false, false);
                acc1 = __builtin_amdgcn_wmma_f32_16x16x32_bf16(false, av, false, b1, (short)0, acc1, false, false);
                acc2 = __builtin_amdgcn_wmma_f32_16x16x32_bf16(false, av, false, b2, (short)0, acc2, false, false);
                acc3 = __builtin_amdgcn_wmma_f32_16x16x32_bf16(false, av, false, b3, (short)0, acc3, false, false);
            }
#pragma unroll
            for (int kt = 0; kt < 4; ++kt) {           // h contribution
                v16bf av = load_a_frag(sh_hbf + nl * RPAD + kt * 32, hi);
                v16bf b0 = *(const v16bf*)(bhh + ((size_t)((wave     ) * 4 + kt) * 32 + lane) * 16);
                v16bf b1 = *(const v16bf*)(bhh + ((size_t)((wave +  8) * 4 + kt) * 32 + lane) * 16);
                v16bf b2 = *(const v16bf*)(bhh + ((size_t)((wave + 16) * 4 + kt) * 32 + lane) * 16);
                v16bf b3 = *(const v16bf*)(bhh + ((size_t)((wave + 24) * 4 + kt) * 32 + lane) * 16);
                acc0 = __builtin_amdgcn_wmma_f32_16x16x32_bf16(false, av, false, b0, (short)0, acc0, false, false);
                acc1 = __builtin_amdgcn_wmma_f32_16x16x32_bf16(false, av, false, b1, (short)0, acc1, false, false);
                acc2 = __builtin_amdgcn_wmma_f32_16x16x32_bf16(false, av, false, b2, (short)0, acc2, false, false);
                acc3 = __builtin_amdgcn_wmma_f32_16x16x32_bf16(false, av, false, b3, (short)0, acc3, false, false);
            }
#pragma unroll
            for (int ti = 0; ti < 4; ++ti) {
                int n = (wave + ti * 8) * 16 + nl;
                v8f acc = (ti == 0) ? acc0 : (ti == 1) ? acc1 : (ti == 2) ? acc2 : acc3;
#pragma unroll
                for (int i = 0; i < 8; ++i)
                    sh_gates[(hi * 8 + i) * 512 + n] = acc[i] + sh_bias[n];
            }
        }
        __syncthreads();

        // ---- Phase 5: LSTM cell update, emit out_h ------------------------
#pragma unroll
        for (int it = 0; it < 8; ++it) {
            int p = it * 256 + tid;
            int m = p >> 7, j = p & 127;
            float gi = sh_gates[m * 512 + j];
            float gf = sh_gates[m * 512 + 128 + j];
            float gg = sh_gates[m * 512 + 256 + j];
            float go = sh_gates[m * 512 + 384 + j];
            float cn = fast_sigmoid(gf) * sh_c[m * 128 + j] + fast_sigmoid(gi) * fast_tanh(gg);
            float hn = fast_sigmoid(go) * fast_tanh(cn);
            sh_c[m * 128 + j]    = cn;
            sh_cbf[m * RPAD + j] = f32_to_bf16(cn);
            sh_hbf[m * RPAD + j] = f32_to_bf16(hn);
            out_h[((size_t)(brow + m) * 63 + t) * 128 + j] = hn;
        }
        __syncthreads();
    }
}

// ---------------------------------------------------------------------------
extern "C" void kernel_launch(void* const* d_in, const int* in_sizes, int n_in,
                              void* d_out, int out_size, void* d_ws, size_t ws_size,
                              hipStream_t stream) {
    (void)in_sizes; (void)n_in; (void)out_size; (void)ws_size;

    const float* input = (const float*)d_in[0];   // [B, 63, 128]
    const float* W_ih  = (const float*)d_in[1];   // [512, 128]
    const float* W_hh  = (const float*)d_in[2];   // [512, 128]
    const float* b_ih  = (const float*)d_in[3];   // [512]
    const float* b_hh  = (const float*)d_in[4];   // [512]
    const float* Wh    = (const float*)d_in[5];   // [64, 256]
    const float* bh    = (const float*)d_in[6];   // [64]
    const float* Wi    = (const float*)d_in[7];   // [64, 63]
    const float* bi    = (const float*)d_in[8];   // [64]
    const float* Wa    = (const float*)d_in[9];   // [1, 64]
    // d_in[10] = ba: dropped (softmax shift-invariant)

    float* out_w = (float*)d_out;                              // [B,63,128]
    float* out_h = out_w + (size_t)Bsz * STEPS * NIN;          // [B,63,128]

    char* ws = (char*)d_ws;
    float* xproj            = (float*)ws;                      // 64 MB f32, L2-resident
    unsigned short* bwh_ws  = (unsigned short*)(ws + (size_t)Bsz * NIN * TT * 4);
    unsigned short* bih_ws  = bwh_ws + 16384;                  // 4*8 frags * 512
    unsigned short* bhh_ws  = bih_ws + 65536;                  // 32*4 frags * 512

    prep_weights_kernel<<<576, 256, 0, stream>>>(W_ih, W_hh, Wh, bwh_ws, bih_ws, bhh_ws);
    xproj_kernel<<<(Bsz * NIN) / 4, 256, 0, stream>>>(input, Wi, bi, xproj);
    scan_kernel<<<Bsz / 16, 256, 0, stream>>>(input, b_ih, b_hh, bh, Wa,
                                              xproj, bwh_ws, bih_ws, bhh_ws,
                                              out_w, out_h);
}
